// DCN_4389456576964
// MI455X (gfx1250) — matrix operand order repaired and checked by reading, not compiled, not run on hardware
//
#include <hip/hip_runtime.h>

// ---------------- types / constants ----------------
typedef __attribute__((ext_vector_type(16))) _Float16 v16h;
typedef __attribute__((ext_vector_type(8)))  float    v8f;
typedef __attribute__((ext_vector_type(4)))  unsigned int v4u;
typedef __attribute__((ext_vector_type(8)))  int      v8i;
typedef __attribute__((ext_vector_type(4)))  int      v4i;

#define KC    576          // Cin*9 contraction length
#define LDSW  584          // padded row stride (f16 elems): 16B-aligned rows, reduced bank conflicts
#define HW    16384        // 128*128
#define NPIX  131072       // 8*128*128
#define NELEM 8388608      // 8*64*128*128

#if __has_builtin(__builtin_amdgcn_tensor_load_to_lds)
#define HAVE_TDM 1
#else
#define HAVE_TDM 0
#endif

#if HAVE_TDM
// Issue one TDM DMA: contiguous copy of nElem 2-byte elements global -> LDS.
// D# packed per CDNA5 ISA (group0/group1; groups 2,3 zero => <=2D).
// 6-arg builtin (clang-23 lane): (v4u g0, v8i g1, v4i g2, v4i g3, v8i extra, i32 cpol).
// Tracked by TENSORcnt; pair with __builtin_amdgcn_s_wait_tensorcnt(0).
__device__ __forceinline__ void tdm_load_to_lds(void* ldsDst, const void* gsrc,
                                                unsigned nElem) {
  unsigned lds = (unsigned)(unsigned long long)ldsDst;          // LDS byte offset (addr[31:0])
  unsigned long long ga = (unsigned long long)gsrc;
  v4u g0;
  g0[0] = 1u;                                                   // count=1, no gather/restore
  g0[1] = lds;                                                  // lds_addr
  g0[2] = (unsigned)(ga & 0xFFFFFFFFu);                         // global_addr[31:0]
  g0[3] = (unsigned)((ga >> 32) & 0x01FFFFFFu) | (2u << 30);    // global_addr[56:32], type=2
  v8i g1;
  g1[0] = (int)(1u << 16);                                      // data_size=1 (2 bytes)
  g1[1] = (int)((nElem & 0xFFFFu) << 16);                       // tensor_dim0[15:0]
  g1[2] = (int)(((nElem >> 16) & 0xFFFFu) | (1u << 16));        // tensor_dim0[31:16], tensor_dim1=1
  g1[3] = (int)((nElem & 0xFFFFu) << 16);                       // tile_dim0 = nElem (fits 16b)
  g1[4] = 1;                                                    // tile_dim1=1, tile_dim2 unused
  g1[5] = (int)nElem;                                           // tensor_dim0_stride[31:0]
  g1[6] = (int)((nElem & 0xFFFFu) << 16);                       // tensor_dim1_stride[15:0]
  g1[7] = (int)(nElem >> 16);                                   // tensor_dim1_stride[47:16]
  v4i g2 = {0, 0, 0, 0};
  v4i g3 = {0, 0, 0, 0};
  v8i g4 = {0, 0, 0, 0, 0, 0, 0, 0};
  __builtin_amdgcn_tensor_load_to_lds(g0, g1, g2, g3, g4, 0);
}
#endif

// A-fragment (16x32 f16): lane holds M=lane&15, K=(lane>>4)*16 + i.  base = row + half*16 + k0.
__device__ __forceinline__ v16h load_a_frag(const _Float16* __restrict__ p) {
  v16h a;
#pragma unroll
  for (int i = 0; i < 16; ++i) a[i] = p[i];
  return a;
}
// B-fragment (32x16 f16): lane holds N=lane&15, K=(lane>>4)*16 + i.
__device__ __forceinline__ v16h load_b_frag(const _Float16* __restrict__ p) {
  v16h b;
#pragma unroll
  for (int i = 0; i < 16; ++i) b[i] = p[i];
  return b;
}

// ---------------- prep kernels ----------------
__global__ __launch_bounds__(256) void cvt_x_kernel(const float* __restrict__ x,
                                                    _Float16* __restrict__ xh) {
  int i = blockIdx.x * 256 + threadIdx.x;
  if (i < NELEM) xh[i] = (_Float16)x[i];
}

// w_dcn [64][576] fp32 -> padded [64][LDSW] f16 staging buffer (TDM source)
__global__ __launch_bounds__(256) void cvt_wd_kernel(const float* __restrict__ wdcn,
                                                     _Float16* __restrict__ wdp) {
  int i = blockIdx.x * 256 + threadIdx.x;
  if (i >= 64 * LDSW) return;
  int o = i / LDSW, kk = i - o * LDSW;
  wdp[i] = (kk < KC) ? (_Float16)wdcn[o * KC + kk] : (_Float16)0.f;
}

// fuse w_off(18) + w_mask(9) -> padded [32][LDSW] f16 + combined bias
__global__ __launch_bounds__(256) void prep_wom_kernel(const float* __restrict__ woff,
                                                       const float* __restrict__ wmask,
                                                       const float* __restrict__ boff,
                                                       const float* __restrict__ bmask,
                                                       _Float16* __restrict__ womp,
                                                       float* __restrict__ bom) {
  int i = blockIdx.x * 256 + threadIdx.x;
  if (i < 32 * LDSW) {
    int r = i / LDSW, kk = i - r * LDSW;
    float v = 0.f;
    if (kk < KC) {
      if (r < 18)      v = woff[r * KC + kk];
      else if (r < 27) v = wmask[(r - 18) * KC + kk];
    }
    womp[i] = (_Float16)v;
  }
  if (i < 32) {
    float b = 0.f;
    if (i < 18)      b = boff[i];
    else if (i < 27) b = bmask[i - 18];
    bom[i] = b;
  }
}

// ---------------- offset/mask conv: implicit GEMM, M=32(27), K=576, N-tile=32 ----------------
__global__ __launch_bounds__(128) void conv_offmask_kernel(const _Float16* __restrict__ xh,
                                                           const _Float16* __restrict__ womp,
                                                           const float* __restrict__ bom,
                                                           float* __restrict__ offm) {
  __shared__ _Float16 Wl[32 * LDSW];            // 36.5 KB weights (TDM-staged)
  __shared__ _Float16 Cl[32 * LDSW];            // 36.5 KB im2col tile
  const int t  = threadIdx.x;
  const int p0 = blockIdx.x * 32;               // 32 pixels, same image, same row
  const int b  = p0 >> 14;
  const int s0 = p0 & (HW - 1);
  const int y  = s0 >> 7;
  const int x0 = s0 & 127;

#if HAVE_TDM
  if (t == 0) tdm_load_to_lds(Wl, womp, 32 * LDSW);   // async: overlaps im2col build
#else
  for (int e = t; e < 32 * LDSW; e += 128) Wl[e] = womp[e];
#endif

  // build im2col tile (zero-padded 3x3 neighborhood, c-major K: ci*9 + ty*3+tx)
  for (int e = t; e < 32 * KC; e += 128) {
    int n  = e / KC, kk = e - n * KC;
    int ci = kk / 9, tt = kk - ci * 9;
    int ty = tt / 3 - 1, tx = (tt % 3) - 1;
    int yy = y + ty, xx = x0 + n + tx;
    _Float16 v = (_Float16)0.f;
    if (yy >= 0 && yy < 128 && xx >= 0 && xx < 128)
      v = xh[((b * 64 + ci) << 14) + (yy << 7) + xx];
    Cl[n * LDSW + kk] = v;
  }
#if HAVE_TDM
  if (t == 0) __builtin_amdgcn_s_wait_tensorcnt(0);
#endif
  __syncthreads();

  const int wave = t >> 5, lane = t & 31;
  const int mt = wave & 1, nt = wave >> 1;      // 2 M-tiles x 2 N-tiles, 4 waves
  const int m = lane & 15, half = lane >> 4;
  v8f acc = {};
  const _Float16* __restrict__ ap = &Wl[(mt * 16 + m) * LDSW + half * 16];
  const _Float16* __restrict__ bp = &Cl[(nt * 16 + m) * LDSW + half * 16];
#pragma unroll 6
  for (int k0 = 0; k0 < KC; k0 += 32) {
    v16h a  = load_a_frag(ap + k0);
    v16h bv = load_b_frag(bp + k0);
    acc = __builtin_amdgcn_wmma_f32_16x16x32_f16(false, a, false, bv, (short)0, acc,
                                                 false, false);
  }

  // epilogue: c<18 -> dy/dx slots, 18..26 -> sigmoid mask, c>=27 dropped
  const int p = p0 + nt * 16 + m;
#pragma unroll
  for (int r = 0; r < 8; ++r) {
    int c = mt * 16 + r + 8 * half;
    if (c >= 27) continue;
    float v = acc[r] + bom[c];
    if (c < 18) {
      int slot = (c & 1) * 9 + (c >> 1);        // dy[k]=slot k, dx[k]=slot 9+k
      offm[p * 28 + slot] = v;
    } else {
      offm[p * 28 + c] = __builtin_amdgcn_rcpf(1.f + __expf(-v));  // sigmoid, v_rcp_f32
    }
  }
}

// ---------------- fused deformable sampling + main GEMM (M=64, K=576, N-tile=32) ----------------
__global__ __launch_bounds__(256) void dcn_sample_gemm_kernel(const float* __restrict__ x,
                                                              const _Float16* __restrict__ wdp,
                                                              const float* __restrict__ offm,
                                                              const float* __restrict__ bdcn,
                                                              float* __restrict__ gout) {
  __shared__ _Float16 Wl[64 * LDSW];            // 73 KB weights (TDM-staged)
  __shared__ _Float16 Cl[32 * LDSW];            // 36.5 KB sampled-column tile
  const int t  = threadIdx.x;
  const int p0 = blockIdx.x * 32;

#if HAVE_TDM
  if (t == 0) tdm_load_to_lds(Wl, wdp, 64 * LDSW);    // async DMA overlaps sampling below
#else
  for (int e = t; e < 64 * LDSW; e += 256) Wl[e] = wdp[e];
#endif

  // ---- fp32 bilinear sampling: thread = (pixel n, channel phase c0) ----
  {
    const int n  = t & 31;
    const int c0 = t >> 5;                      // 0..7, handles channels c0+8*i
    const int p  = p0 + n;
    const int b  = p >> 14;
    const int s  = p & (HW - 1);
    const int yi = s >> 7, xi = s & 127;
    const float* __restrict__ xb = x + ((long)b << 20);   // b*64*HW
#pragma unroll
    for (int k = 0; k < 9; ++k) {
      float dy = offm[p * 28 + k];
      float dx = offm[p * 28 + 9 + k];
      float mk = offm[p * 28 + 18 + k];
      float py = (float)yi + (float)(k / 3 - 1) + dy;
      float px = (float)xi + (float)(k % 3 - 1) + dx;
      float fy = floorf(py), fx = floorf(px);
      float ly = py - fy, lx = px - fx;
      int y0 = (int)fy, x0 = (int)fx, y1 = y0 + 1, x1 = x0 + 1;
      float vy0 = (y0 >= 0 && y0 < 128) ? 1.f : 0.f;
      float vy1 = (y1 >= 0 && y1 < 128) ? 1.f : 0.f;
      float vx0 = (x0 >= 0 && x0 < 128) ? 1.f : 0.f;
      float vx1 = (x1 >= 0 && x1 < 128) ? 1.f : 0.f;
      int yc0 = min(max(y0, 0), 127), yc1 = min(max(y1, 0), 127);
      int xc0 = min(max(x0, 0), 127), xc1 = min(max(x1, 0), 127);
      float w00 = (1.f - ly) * (1.f - lx) * vy0 * vx0;
      float w01 = (1.f - ly) * lx * vy0 * vx1;
      float w10 = ly * (1.f - lx) * vy1 * vx0;
      float w11 = ly * lx * vy1 * vx1;
      int o00 = (yc0 << 7) + xc0, o01 = (yc0 << 7) + xc1;
      int o10 = (yc1 << 7) + xc0, o11 = (yc1 << 7) + xc1;
#pragma unroll
      for (int i = 0; i < 8; ++i) {
        int c = c0 + (i << 3);
        const float* __restrict__ xc = xb + (c << 14);
        float v = xc[o00] * w00 + xc[o01] * w01 + xc[o10] * w10 + xc[o11] * w11;
        Cl[n * LDSW + c * 9 + k] = (_Float16)(v * mk);
      }
    }
  }
#if HAVE_TDM
  if (t == 0) __builtin_amdgcn_s_wait_tensorcnt(0);
#endif
  __syncthreads();

  // ---- WMMA: 8 waves = 4 M-tiles x 2 N-tiles, 18 K-steps ----
  const int wave = t >> 5, lane = t & 31;
  const int mt = wave & 3, nt = wave >> 2;
  const int m = lane & 15, half = lane >> 4;
  v8f acc = {};
  const _Float16* __restrict__ ap = &Wl[(mt * 16 + m) * LDSW + half * 16];
  const _Float16* __restrict__ bp = &Cl[(nt * 16 + m) * LDSW + half * 16];
#pragma unroll 6
  for (int k0 = 0; k0 < KC; k0 += 32) {
    v16h a  = load_a_frag(ap + k0);
    v16h bv = load_b_frag(bp + k0);
    acc = __builtin_amdgcn_wmma_f32_16x16x32_f16(false, a, false, bv, (short)0, acc,
                                                 false, false);
  }

  const int p = p0 + nt * 16 + m;
#pragma unroll
  for (int r = 0; r < 8; ++r) {
    int c = mt * 16 + r + 8 * half;
    gout[c * NPIX + p] = acc[r] + bdcn[c];      // channel-major for BN reduction
  }
}

// ---------------- batchnorm statistics: one block per channel ----------------
__global__ __launch_bounds__(256) void bn_stats_kernel(const float* __restrict__ gout,
                                                       float* __restrict__ stats) {
  const int c = blockIdx.x;
  const float* __restrict__ g = gout + (long)c * NPIX;
  float s1 = 0.f, s2 = 0.f;
  for (int i = threadIdx.x; i < NPIX; i += 256) {
    float v = g[i];
    s1 += v; s2 += v * v;
  }
  __shared__ float a1[256], a2[256];
  a1[threadIdx.x] = s1; a2[threadIdx.x] = s2;
  __syncthreads();
  for (int off = 128; off > 0; off >>= 1) {
    if (threadIdx.x < off) {
      a1[threadIdx.x] += a1[threadIdx.x + off];
      a2[threadIdx.x] += a2[threadIdx.x + off];
    }
    __syncthreads();
  }
  if (threadIdx.x == 0) {
    float mean = a1[0] * (1.f / (float)NPIX);
    float var  = a2[0] * (1.f / (float)NPIX) - mean * mean;
    stats[2 * c]     = mean;
    stats[2 * c + 1] = rsqrtf(var + 1e-5f);
  }
}

// ---------------- BN apply + Mish, write NCHW output ----------------
__global__ __launch_bounds__(256) void bn_mish_kernel(const float* __restrict__ gout,
                                                      const float* __restrict__ stats,
                                                      const float* __restrict__ gamma,
                                                      const float* __restrict__ beta,
                                                      float* __restrict__ out) {
  int idx = blockIdx.x * 256 + threadIdx.x;
  if (idx >= NELEM) return;
  int b = idx >> 20;                 // /(64*HW)
  int o = (idx >> 14) & 63;
  int s = idx & (HW - 1);
  float v = gout[o * NPIX + (b << 14) + s];
  v = (v - stats[2 * o]) * stats[2 * o + 1] * gamma[o] + beta[o];
  float sp = fmaxf(v, 0.f) + log1pf(__expf(-fabsf(v)));   // stable softplus
  out[idx] = v * tanhf(sp);
}

// ---------------- launch ----------------
extern "C" void kernel_launch(void* const* d_in, const int* in_sizes, int n_in,
                              void* d_out, int out_size, void* d_ws, size_t ws_size,
                              hipStream_t stream) {
  (void)in_sizes; (void)n_in; (void)out_size; (void)ws_size;
  const float* x      = (const float*)d_in[0];
  const float* w_off  = (const float*)d_in[1];
  const float* b_off  = (const float*)d_in[2];
  const float* w_mask = (const float*)d_in[3];
  const float* b_mask = (const float*)d_in[4];
  const float* w_dcn  = (const float*)d_in[5];
  const float* b_dcn  = (const float*)d_in[6];
  const float* gamma  = (const float*)d_in[7];
  const float* beta   = (const float*)d_in[8];
  float* out = (float*)d_out;

  char* ws = (char*)d_ws;
  size_t off = 0;
  auto alloc = [&](size_t bytes) -> char* {
    char* p = ws + off;
    off = (off + bytes + 255) & ~(size_t)255;
    return p;
  };
  _Float16* xh    = (_Float16*)alloc((size_t)NELEM * 2);          // 16.8 MB
  _Float16* wdp   = (_Float16*)alloc((size_t)64 * LDSW * 2);      // 74.8 KB (padded, TDM src)
  _Float16* womp  = (_Float16*)alloc((size_t)32 * LDSW * 2);      // 37.4 KB (padded, TDM src)
  float*    bom   = (float*)alloc(32 * sizeof(float));
  float*    offm  = (float*)alloc((size_t)NPIX * 28 * 4);         // 14.7 MB
  float*    gout  = (float*)alloc((size_t)64 * NPIX * 4);         // 33.6 MB
  float*    stats = (float*)alloc(128 * sizeof(float));

  cvt_x_kernel<<<NELEM / 256, 256, 0, stream>>>(x, xh);
  cvt_wd_kernel<<<(64 * LDSW + 255) / 256, 256, 0, stream>>>(w_dcn, wdp);
  prep_wom_kernel<<<(32 * LDSW + 255) / 256, 256, 0, stream>>>(w_off, w_mask, b_off, b_mask,
                                                               womp, bom);
  conv_offmask_kernel<<<NPIX / 32, 128, 0, stream>>>(xh, womp, bom, offm);
  dcn_sample_gemm_kernel<<<NPIX / 32, 256, 0, stream>>>(x, wdp, offm, b_dcn, gout);
  bn_stats_kernel<<<64, 256, 0, stream>>>(gout, stats);
  bn_mish_kernel<<<NELEM / 256, 256, 0, stream>>>(gout, stats, gamma, beta, out);
}
